// Quantize_2156073583342
// MI455X (gfx1250) — matrix operand order repaired and checked by reading, not compiled, not run on hardware
//
#include <hip/hip_runtime.h>
#include <hip/hip_bf16.h>

typedef __attribute__((ext_vector_type(16))) _Float16 v16h;
typedef __attribute__((ext_vector_type(8)))  _Float16 v8h;
typedef __attribute__((ext_vector_type(8)))  float    v8f;

#define CODES 1024
#define CDIM  256
#define HW    4096            // 64*64
#define BATCH 16
#define NPOS  (BATCH * HW)    // 65536 positions
#define LDSP  264             // padded LDS row stride (halves)
#define POS_PER_WG 128        // 8 waves x 16 positions
#define GCODES 128            // codes per staged group
#define GROUPS (CODES / GCODES)

// ---------------------------------------------------------------------------
// Prep: weight f32 -> f16 (WMMA A operand), plus per-code squared L2 norm.
// ---------------------------------------------------------------------------
__global__ __launch_bounds__(256) void vq_prep(const float* __restrict__ w,
                                               _Float16* __restrict__ wf16,
                                               float* __restrict__ wsum) {
  __shared__ float red[256];
  const int n = blockIdx.x;
  const int t = threadIdx.x;
  float v = w[n * CDIM + t];
  wf16[n * CDIM + t] = (_Float16)v;
  red[t] = v * v;
  __syncthreads();
  for (int s = 128; s > 0; s >>= 1) {
    if (t < s) red[t] += red[t + s];
    __syncthreads();
  }
  if (t == 0) wsum[n] = red[0];
}

// order-preserving float -> u32 key (handles negatives)
__device__ inline unsigned fkey(float f) {
  unsigned u = __float_as_uint(f);
  return (u & 0x80000000u) ? ~u : (u | 0x80000000u);
}

// ---------------------------------------------------------------------------
// Main: one workgroup = 128 positions vs all 1024 codes.
// Codebook groups double-buffered in LDS via global_load_async_to_lds_b128:
// group g+1 is in flight while group g feeds 64 WMMAs per wave.
// ---------------------------------------------------------------------------
__global__ __launch_bounds__(256) void vq_main(const float* __restrict__ z,
                                               const float* __restrict__ w,
                                               const _Float16* __restrict__ wf16,
                                               const float* __restrict__ wsum,
                                               float* __restrict__ q_out,
                                               float* __restrict__ st_out,
                                               float* __restrict__ idx_out) {
  __shared__ _Float16 lds_z[POS_PER_WG * LDSP];        // z tile [pos][chan] f16
  __shared__ _Float16 lds_w[2 * GCODES * LDSP];        // double-buffered codes
  __shared__ float lds_wsum[CODES];                    // ||w||^2
  __shared__ unsigned long long slot[POS_PER_WG];      // packed (key<<32)|code

  const int t    = threadIdx.x;
  const int lane = t & 31;
  const int wave = t >> 5;                 // 8 waves
  const int n0   = blockIdx.x * POS_PER_WG;
  const int b    = n0 >> 12;
  const int hw0  = n0 & (HW - 1);
  const int p    = t & 15;
  const int crow = t >> 4;                 // 0..15

  // this thread's 16 async chunks (8 halves each) of a 128x256 group
  const int s_code = (t * 16) >> 5;        // reused below via chunk math

  // ---- kick off async fill of group 0 into buffer 0 ----
  {
    const _Float16* wsrc = wf16;
#pragma unroll
    for (int i = 0; i < 16; ++i) {
      int chunk = t + i * 256;                         // 0..4095
      int code = chunk >> 5;
      int cc = chunk & 31;
      unsigned dst = (unsigned)(uintptr_t)&lds_w[code * LDSP + cc * 8];
      unsigned long long src =
          (unsigned long long)(uintptr_t)(wsrc + code * CDIM + cc * 8);
      asm volatile("global_load_async_to_lds_b128 %0, %1, off"
                   :: "v"(dst), "v"(src) : "memory");
    }
  }

  // ---- overlap: stage z tile (128 pos x 256 ch) to LDS as f16 ----
  const float* zb = z + (size_t)b * (CDIM * HW) + hw0;
  for (int sg = 0; sg < 8; ++sg) {
    int pos = sg * 16 + p;
#pragma unroll
    for (int i = 0; i < 16; ++i) {
      int c = crow + i * 16;
      lds_z[pos * LDSP + c] = (_Float16)zb[(size_t)c * HW + pos];
    }
  }
  for (int i = t; i < CODES; i += 256) lds_wsum[i] = wsum[i];
  if (t < POS_PER_WG) slot[t] = ~0ull;

  asm volatile("s_wait_asynccnt 0x0" ::: "memory");
  __syncthreads();                                     // group 0 + z ready

  const int lp = lane & 31 & 15;
  const int hi = (lane >= 16) ? 1 : 0;
  const int zrow = (wave * 16 + (lane & 15)) * LDSP;   // this lane's z row

  unsigned long long best = ~0ull;

  for (int g = 0; g < GROUPS; ++g) {
    // ---- issue async fill of the NEXT group into the other buffer ----
    if (g + 1 < GROUPS) {
      const _Float16* wsrc = wf16 + (size_t)(g + 1) * GCODES * CDIM;
      _Float16* dbuf = lds_w + ((g + 1) & 1) * (GCODES * LDSP);
#pragma unroll
      for (int i = 0; i < 16; ++i) {
        int chunk = t + i * 256;
        int code = chunk >> 5;
        int cc = chunk & 31;
        unsigned dst = (unsigned)(uintptr_t)&dbuf[code * LDSP + cc * 8];
        unsigned long long src =
            (unsigned long long)(uintptr_t)(wsrc + code * CDIM + cc * 8);
        asm volatile("global_load_async_to_lds_b128 %0, %1, off"
                     :: "v"(dst), "v"(src) : "memory");
      }
    }

    // ---- GEMM over current group: 8 M-tiles x 1 N-tile, K = 256 ----
    const _Float16* wt = lds_w + (g & 1) * (GCODES * LDSP);
    v8f acc[8];
#pragma unroll
    for (int i = 0; i < 8; ++i) acc[i] = (v8f){0.f,0.f,0.f,0.f,0.f,0.f,0.f,0.f};

    for (int kk = 0; kk < CDIM; kk += 32) {
      union { v16h v; v8h h[2]; } bu;
      const _Float16* bsrc = &lds_z[zrow + kk + (hi ? 8 : 0)];
      bu.h[0] = *(const v8h*)(bsrc);                   // ds_load_b128
      bu.h[1] = *(const v8h*)(bsrc + 16);
#pragma unroll
      for (int ti = 0; ti < 8; ++ti) {
        const _Float16* arow = &wt[(ti * 16 + lp) * LDSP + kk + (hi ? 8 : 0)];
        union { v16h v; v8h h[2]; } au;
        au.h[0] = *(const v8h*)(arow);                 // ds_load_b128
        au.h[1] = *(const v8h*)(arow + 16);
        acc[ti] = __builtin_amdgcn_wmma_f32_16x16x32_f16(
            false, au.v, false, bu.v, (short)0, acc[ti], false, false);
      }
    }

    // ---- fold distances into running packed argmin ----
#pragma unroll
    for (int ti = 0; ti < 8; ++ti) {
#pragma unroll
      for (int r = 0; r < 8; ++r) {
        int code = g * GCODES + ti * 16 + r + (hi ? 8 : 0);
        float d = lds_wsum[code] - 2.0f * acc[ti][r];
        unsigned long long pk =
            ((unsigned long long)fkey(d) << 32) | (unsigned)code;
        best = pk < best ? pk : best;
      }
    }

    // next group's fill must be complete before anyone reads it
    asm volatile("s_wait_asynccnt 0x0" ::: "memory");
    __syncthreads();
  }

  atomicMin(&slot[wave * 16 + lp], best);              // merge the two half-lanes
  __syncthreads();

  // ---- write outputs ----
  if (t < POS_PER_WG)
    idx_out[n0 + t] = (float)(int)(slot[t] & 0xFFFFFFFFull);

#pragma unroll
  for (int sg = 0; sg < 8; ++sg) {
    int pos = sg * 16 + p;
    int idx = (int)(slot[pos] & 0xFFFFFFFFull);
    const float* wrow = w + (size_t)idx * CDIM;
    const size_t obase = (size_t)b * (CDIM * HW) + hw0 + pos;
#pragma unroll
    for (int i = 0; i < 16; ++i) {
      int c = crow + i * 16;
      float v = wrow[c];
      size_t o = obase + (size_t)c * HW;
      q_out[o]  = v;   // quantized
      st_out[o] = v;   // straight_through == quantized in forward
    }
  }
  (void)s_code;
}

// ---------------------------------------------------------------------------
extern "C" void kernel_launch(void* const* d_in, const int* in_sizes, int n_in,
                              void* d_out, int out_size, void* d_ws, size_t ws_size,
                              hipStream_t stream) {
  const float* z = (const float*)d_in[0];   // [16,256,64,64]
  const float* w = (const float*)d_in[1];   // [1024,256]

  // workspace: f16 weight (512 KB) + per-code norms (4 KB)
  _Float16* wf16 = (_Float16*)d_ws;
  float* wsumb = (float*)((char*)d_ws + (size_t)CODES * CDIM * sizeof(_Float16));

  float* q_out  = (float*)d_out;
  float* st_out = q_out + (size_t)BATCH * CDIM * HW;
  float* idx_out = st_out + (size_t)BATCH * CDIM * HW;

  vq_prep<<<CODES, 256, 0, stream>>>(w, wf16, wsumb);
  vq_main<<<NPOS / POS_PER_WG, 256, 0, stream>>>(z, w, wf16, wsumb,
                                                 q_out, st_out, idx_out);
}
